// EGNNEncoder_57002805953146
// MI455X (gfx1250) — matrix operand order repaired and checked by reading, not compiled, not run on hardware
//
#include <hip/hip_runtime.h>
#include <hip/hip_bf16.h>

#define NN 50000
#define NE 600000
#define IN_DIM 16
#define HID 128
#define NLAYERS 3

typedef __attribute__((ext_vector_type(16))) __bf16 v16bf;
typedef __attribute__((ext_vector_type(8)))  float  v8f;

// ---------------- device helpers ----------------

__device__ __forceinline__ unsigned short f2bf(float f) {
  union { float f; unsigned int u; } a; a.f = f;
  unsigned int u = a.u;
  unsigned int r = u + 0x7fffu + ((u >> 16) & 1u);   // round-to-nearest-even
  return (unsigned short)(r >> 16);
}

__device__ __forceinline__ float silu(float x) { return x / (1.0f + __expf(-x)); }

// A-fragment (16x32 bf16, MxK) from an LDS row-major bf16 tile.
// ISA layout: lane M = lane&15; VGPR0..3 hold K = kb+0..7 (kb = lane>=16 ? 8 : 0),
// VGPR4..7 hold K = 16+kb+0..7.  v16bf element i -> VGPR i/2.
__device__ __forceinline__ v16bf load_a_frag(const unsigned short* s, int lane,
                                             int kbase, int ld) {
  union { v16bf v; unsigned int u[8]; } f;
  const int m  = lane & 15;
  const int kb = kbase + ((lane >= 16) ? 8 : 0);
  const unsigned short* row = s + m * ld + kb;
#pragma unroll
  for (int i = 0; i < 4; ++i) f.u[i]     = *(const unsigned int*)(row + 2 * i);
#pragma unroll
  for (int i = 0; i < 4; ++i) f.u[4 + i] = *(const unsigned int*)(row + 16 + 2 * i);
  return f.v;
}

__device__ __forceinline__ v8f wmma_bf16(v16bf a, v16bf b, v8f c) {
  return __builtin_amdgcn_wmma_f32_16x16x32_bf16(false, a, false, b, (short)0, c,
                                                 false, false);
}

// Async global->LDS copy of one 16B chunk per lane (CDNA5, tracked by ASYNCcnt).
// dst_lds = LDS byte address (low 32 bits of generic shared pointer).
__device__ __forceinline__ void async_ldst16(unsigned dst_lds, const void* src) {
  asm volatile("global_load_async_to_lds_b128 %0, %1, off"
               :: "v"(dst_lds), "v"(src) : "memory");
}
__device__ __forceinline__ void async_wait0() {
  asm volatile("s_wait_asynccnt 0x0" ::: "memory");
}

// ---------------- weight pre-pack (B-fragment order) ----------------
// B (32x16 bf16, KxN) layout: lane N = lane&15; element i -> K = (i&7) + (lane>=16?8:0) + (i>=8?16:0).
// Packed as blocks of (kstep, ntile): 32 lanes x 16 halves contiguous per lane.
__global__ void pack_b_kernel(const float* __restrict__ W, unsigned short* __restrict__ out,
                              int K, int N) {
  int idx = blockIdx.x * blockDim.x + threadIdx.x;
  int total = (K >> 5) * (N >> 4) * 512;
  if (idx >= total) return;
  int lanei = idx & 511;
  int lane = lanei >> 4;
  int i    = lanei & 15;
  int blk  = idx >> 9;
  int nt = blk % (N >> 4);
  int kt = blk / (N >> 4);
  int k = kt * 32 + (i & 7) + ((lane >= 16) ? 8 : 0) + ((i >= 8) ? 16 : 0);
  int n = nt * 16 + (lane & 15);
  out[idx] = f2bf(W[(size_t)k * N + n]);
}

// ---------------- small utility kernels ----------------

__global__ void zero_kernel(float* p, int n) {
  int i = blockIdx.x * blockDim.x + threadIdx.x;
  if (i < n) p[i] = 0.0f;
}

__global__ void copy_kernel(const float* __restrict__ src, float* __restrict__ dst, int n) {
  int i = blockIdx.x * blockDim.x + threadIdx.x;
  if (i < n) dst[i] = src[i];
}

// h = silu(x @ proj_w + proj_b); also writes bf16 mirror of h.
__global__ __launch_bounds__(128) void proj_kernel(const float* __restrict__ x,
                                                   const float* __restrict__ w,
                                                   const float* __restrict__ b,
                                                   float* __restrict__ hout,
                                                   unsigned short* __restrict__ hbf) {
  const int i = blockIdx.x;
  const int o = threadIdx.x;
  float acc = b[o];
#pragma unroll
  for (int k = 0; k < IN_DIM; ++k) acc = fmaf(x[(size_t)i * IN_DIM + k], w[k * HID + o], acc);
  float v = silu(acc);
  hout[(size_t)i * HID + o] = v;
  hbf[(size_t)i * HID + o] = f2bf(v);
}

__global__ void coord_update_kernel(float* __restrict__ coords,
                                    const float* __restrict__ xagg,
                                    const float* __restrict__ cnt, int n) {
  int i = blockIdx.x * blockDim.x + threadIdx.x;
  if (i >= n) return;
  float c = fmaxf(cnt[i], 1.0f);
  coords[2 * i]     += xagg[2 * i]     / c;
  coords[2 * i + 1] += xagg[2 * i + 1] / c;
}

// ---------------- edge kernel: 16 edges per workgroup (4 waves) ----------------

__global__ __launch_bounds__(128) void egnn_edge_kernel(
    const unsigned short* __restrict__ hbf, const float* __restrict__ coords,
    const long long* __restrict__ erow, const long long* __restrict__ ecol,
    const unsigned short* __restrict__ pk_ew1, const float* __restrict__ eb1,
    const float* __restrict__ ew1_dist,
    const unsigned short* __restrict__ pk_ew2, const float* __restrict__ eb2,
    const unsigned short* __restrict__ pk_cw1, const float* __restrict__ cb1,
    const float* __restrict__ cw2, const float* __restrict__ cb2,
    float* __restrict__ msg_agg, float* __restrict__ xagg, float* __restrict__ cnt,
    int updateCoords) {
  __shared__ __align__(16) unsigned short sA[16 * 264];    // e_in tile bf16 [16 x 256]
  __shared__ __align__(16) unsigned short sMid[16 * 136];  // after GEMM1+silu
  __shared__ __align__(16) unsigned short sM[16 * 136];    // message m (bf16)
  __shared__ __align__(16) float sT[16 * 132];             // coord-MLP hidden (f32)
  __shared__ int   sRow[16], sCol[16];
  __shared__ float sDist[16], sDx[16], sDy[16];

  const int t    = threadIdx.x;
  const int lane = t & 31;
  const int wave = t >> 5;
  const int eb   = blockIdx.x * 16;

  if (t < 16) {
    int r = (int)erow[eb + t];
    int c = (int)ecol[eb + t];
    sRow[t] = r; sCol[t] = c;
    float dx = coords[2 * r] - coords[2 * c];
    float dy = coords[2 * r + 1] - coords[2 * c + 1];
    sDx[t] = dx; sDy[t] = dy;
    sDist[t] = dx * dx + dy * dy;
  }
  __syncthreads();

  { // async gather h_bf16[row] || h_bf16[col] -> sA tile (8KB, 512x16B chunks)
    const unsigned sA_base = (unsigned)(size_t)(void*)sA;
#pragma unroll
    for (int part = 0; part < 4; ++part) {
      int c = part * 128 + t;        // chunk id 0..511
      int e = c >> 5;                // edge within tile
      int j = c & 31;                // 0..15: row half, 16..31: col half
      int jj = j & 15;
      int node = (j < 16) ? sRow[e] : sCol[e];
      const unsigned short* src = hbf + (size_t)node * HID + jj * 8;
      unsigned dst = sA_base + (unsigned)(e * 264 + ((j < 16) ? 0 : 128) + jj * 8) * 2u;
      async_ldst16(dst, (const void*)src);
    }
    async_wait0();
  }
  __syncthreads();

  const int ncol = lane & 15;
  const int mhi  = (lane >= 16) ? 8 : 0;
  const int n0   = (2 * wave) * 16 + ncol;
  const int n1   = n0 + 16;

  { // GEMM1: e_in[16x256] @ e_w1 (+dist rank-1) -> silu -> sMid
    v8f a0 = {0,0,0,0,0,0,0,0}, a1 = {0,0,0,0,0,0,0,0};
    const v16bf* B = (const v16bf*)pk_ew1;
#pragma unroll
    for (int ks = 0; ks < 8; ++ks) {
      v16bf a  = load_a_frag(sA, lane, ks * 32, 264);
      v16bf b0 = B[(ks * 8 + 2 * wave) * 32 + lane];
      v16bf b1 = B[(ks * 8 + 2 * wave + 1) * 32 + lane];
      a0 = wmma_bf16(a, b0, a0);
      a1 = wmma_bf16(a, b1, a1);
    }
    float bias0 = eb1[n0], bias1 = eb1[n1];
    float w0 = ew1_dist[n0], w1 = ew1_dist[n1];
#pragma unroll
    for (int r = 0; r < 8; ++r) {
      int m = r + mhi;
      float d = sDist[m];
      sMid[m * 136 + n0] = f2bf(silu(a0[r] + bias0 + d * w0));
      sMid[m * 136 + n1] = f2bf(silu(a1[r] + bias1 + d * w1));
    }
  }
  __syncthreads();

  { // GEMM2: sMid @ e_w2 -> silu -> m; scatter-add into msg_agg[row]
    v8f a0 = {0,0,0,0,0,0,0,0}, a1 = {0,0,0,0,0,0,0,0};
    const v16bf* B = (const v16bf*)pk_ew2;
#pragma unroll
    for (int ks = 0; ks < 4; ++ks) {
      v16bf a  = load_a_frag(sMid, lane, ks * 32, 136);
      v16bf b0 = B[(ks * 8 + 2 * wave) * 32 + lane];
      v16bf b1 = B[(ks * 8 + 2 * wave + 1) * 32 + lane];
      a0 = wmma_bf16(a, b0, a0);
      a1 = wmma_bf16(a, b1, a1);
    }
    float bias0 = eb2[n0], bias1 = eb2[n1];
#pragma unroll
    for (int r = 0; r < 8; ++r) {
      int m = r + mhi;
      float v0 = silu(a0[r] + bias0);
      float v1 = silu(a1[r] + bias1);
      sM[m * 136 + n0] = f2bf(v0);
      sM[m * 136 + n1] = f2bf(v1);
      float* dst = msg_agg + (size_t)sRow[m] * HID;
      unsafeAtomicAdd(dst + n0, v0);
      unsafeAtomicAdd(dst + n1, v1);
    }
  }

  if (updateCoords) {
    __syncthreads();
    { // GEMM3: m @ c_w1 -> silu -> sT (f32)
      v8f a0 = {0,0,0,0,0,0,0,0}, a1 = {0,0,0,0,0,0,0,0};
      const v16bf* B = (const v16bf*)pk_cw1;
#pragma unroll
      for (int ks = 0; ks < 4; ++ks) {
        v16bf a  = load_a_frag(sM, lane, ks * 32, 136);
        v16bf b0 = B[(ks * 8 + 2 * wave) * 32 + lane];
        v16bf b1 = B[(ks * 8 + 2 * wave + 1) * 32 + lane];
        a0 = wmma_bf16(a, b0, a0);
        a1 = wmma_bf16(a, b1, a1);
      }
      float bias0 = cb1[n0], bias1 = cb1[n1];
#pragma unroll
      for (int r = 0; r < 8; ++r) {
        int m = r + mhi;
        sT[m * 132 + n0] = silu(a0[r] + bias0);
        sT[m * 132 + n1] = silu(a1[r] + bias1);
      }
    }
    __syncthreads();
    if (t < 16) { // cw = t @ c_w2 + c_b2 ; scatter coord updates
      float acc = cb2[0];
      const float* row = sT + t * 132;
#pragma unroll 4
      for (int j = 0; j < HID; ++j) acc = fmaf(row[j], cw2[j], acc);
      int r = sRow[t];
      unsafeAtomicAdd(xagg + 2 * r,     sDx[t] * acc);
      unsafeAtomicAdd(xagg + 2 * r + 1, sDy[t] * acc);
      unsafeAtomicAdd(cnt + r, 1.0f);
    }
  }
}

// ---------------- node kernel: 16 nodes per workgroup ----------------

__global__ __launch_bounds__(128) void egnn_node_kernel(
    const float* __restrict__ h, const unsigned short* __restrict__ hbf_in,
    const float* __restrict__ msg_agg,
    const unsigned short* __restrict__ pk_nw1, const float* __restrict__ nb1,
    const unsigned short* __restrict__ pk_nw2, const float* __restrict__ nb2,
    const float* __restrict__ lng, const float* __restrict__ lnb,
    float* __restrict__ hout, unsigned short* __restrict__ hbf_out) {
  __shared__ __align__(16) unsigned short sA[16 * 264];
  __shared__ __align__(16) unsigned short sMid[16 * 136];
  __shared__ __align__(16) float sT[16 * 132];
  __shared__ float sMu[16], sRstd[16];

  const int t     = threadIdx.x;
  const int lane  = t & 31;
  const int wave  = t >> 5;
  const int nbase = blockIdx.x * 16;

  { // node_in = h(bf16 mirror) || cvt(msg_agg) -> bf16 tile
    int e = t >> 3, seg = t & 7;
    size_t node = (size_t)(nbase + e);
    const uint4* hr = (const uint4*)(hbf_in + node * HID) + seg * 2;  // 16 halves
    const float4* mr = (const float4*)(msg_agg + node * HID) + seg * 4;
    uint4* dH = (uint4*)(sA + e * 264 + seg * 16);
    unsigned short* dM = sA + e * 264 + 128 + seg * 16;
    dH[0] = hr[0];
    dH[1] = hr[1];
#pragma unroll
    for (int q = 0; q < 4; ++q) {
      float4 b = mr[q];
      dM[4 * q + 0] = f2bf(b.x); dM[4 * q + 1] = f2bf(b.y);
      dM[4 * q + 2] = f2bf(b.z); dM[4 * q + 3] = f2bf(b.w);
    }
  }
  __syncthreads();

  const int ncol = lane & 15;
  const int mhi  = (lane >= 16) ? 8 : 0;
  const int n0   = (2 * wave) * 16 + ncol;
  const int n1   = n0 + 16;

  { // GEMM1: node_in @ n_w1 -> silu -> sMid
    v8f a0 = {0,0,0,0,0,0,0,0}, a1 = {0,0,0,0,0,0,0,0};
    const v16bf* B = (const v16bf*)pk_nw1;
#pragma unroll
    for (int ks = 0; ks < 8; ++ks) {
      v16bf a  = load_a_frag(sA, lane, ks * 32, 264);
      v16bf b0 = B[(ks * 8 + 2 * wave) * 32 + lane];
      v16bf b1 = B[(ks * 8 + 2 * wave + 1) * 32 + lane];
      a0 = wmma_bf16(a, b0, a0);
      a1 = wmma_bf16(a, b1, a1);
    }
    float bias0 = nb1[n0], bias1 = nb1[n1];
#pragma unroll
    for (int r = 0; r < 8; ++r) {
      int m = r + mhi;
      sMid[m * 136 + n0] = f2bf(silu(a0[r] + bias0));
      sMid[m * 136 + n1] = f2bf(silu(a1[r] + bias1));
    }
  }
  __syncthreads();

  { // GEMM2: sMid @ n_w2 + bias + residual h -> sT (f32)
    v8f a0 = {0,0,0,0,0,0,0,0}, a1 = {0,0,0,0,0,0,0,0};
    const v16bf* B = (const v16bf*)pk_nw2;
#pragma unroll
    for (int ks = 0; ks < 4; ++ks) {
      v16bf a  = load_a_frag(sMid, lane, ks * 32, 136);
      v16bf b0 = B[(ks * 8 + 2 * wave) * 32 + lane];
      v16bf b1 = B[(ks * 8 + 2 * wave + 1) * 32 + lane];
      a0 = wmma_bf16(a, b0, a0);
      a1 = wmma_bf16(a, b1, a1);
    }
    float bias0 = nb2[n0], bias1 = nb2[n1];
#pragma unroll
    for (int r = 0; r < 8; ++r) {
      int m = r + mhi;
      size_t node = (size_t)(nbase + m);
      sT[m * 132 + n0] = a0[r] + bias0 + h[node * HID + n0];
      sT[m * 132 + n1] = a1[r] + bias1 + h[node * HID + n1];
    }
  }
  __syncthreads();

  if (t < 16) { // per-row layernorm stats
    const float* row = sT + t * 132;
    float mu = 0.0f;
#pragma unroll 4
    for (int j = 0; j < HID; ++j) mu += row[j];
    mu *= (1.0f / HID);
    float var = 0.0f;
#pragma unroll 4
    for (int j = 0; j < HID; ++j) { float d = row[j] - mu; var = fmaf(d, d, var); }
    var *= (1.0f / HID);
    sMu[t] = mu;
    sRstd[t] = rsqrtf(var + 1e-5f);
  }
  __syncthreads();

  { // normalize + affine -> f32 output + bf16 mirror for next layer
    int e = t >> 3, seg = t & 7;
    float mu = sMu[e], rs = sRstd[e];
    size_t node = (size_t)(nbase + e);
#pragma unroll
    for (int j = 0; j < 16; ++j) {
      int c = seg * 16 + j;
      float v = (sT[e * 132 + c] - mu) * rs * lng[c] + lnb[c];
      hout[node * HID + c] = v;
      hbf_out[node * HID + c] = f2bf(v);
    }
  }
}

// ---------------- host ----------------

extern "C" void kernel_launch(void* const* d_in, const int* in_sizes, int n_in,
                              void* d_out, int out_size, void* d_ws, size_t ws_size,
                              hipStream_t stream) {
  (void)in_sizes; (void)n_in; (void)out_size; (void)ws_size;

  const float*     x      = (const float*)d_in[0];
  const float*     pos    = (const float*)d_in[1];
  const long long* ei     = (const long long*)d_in[2];   // int64 [2, NE]
  const float*     proj_w = (const float*)d_in[3];
  const float*     proj_b = (const float*)d_in[4];

  // params['layers'] flattened in dict insertion order; layers 0/1 have coord MLP.
  const int base[NLAYERS] = {5, 19, 33};
  const float *ew1[3], *eb1[3], *ew2[3], *eb2[3];
  const float *cw1[2], *cb1[2], *cw2[2], *cb2[2];
  const float *nw1[3], *nb1[3], *nw2[3], *nb2[3], *lng[3], *lnb[3];
  for (int l = 0; l < NLAYERS; ++l) {
    int b = base[l];
    ew1[l] = (const float*)d_in[b + 0];
    eb1[l] = (const float*)d_in[b + 1];
    ew2[l] = (const float*)d_in[b + 2];
    eb2[l] = (const float*)d_in[b + 3];
    int o = 4;
    if (l < 2) {
      cw1[l] = (const float*)d_in[b + 4];
      cb1[l] = (const float*)d_in[b + 5];
      cw2[l] = (const float*)d_in[b + 6];
      cb2[l] = (const float*)d_in[b + 7];
      o = 8;
    }
    nw1[l] = (const float*)d_in[b + o + 0];
    nb1[l] = (const float*)d_in[b + o + 1];
    nw2[l] = (const float*)d_in[b + o + 2];
    nb2[l] = (const float*)d_in[b + o + 3];
    lng[l] = (const float*)d_in[b + o + 4];
    lnb[l] = (const float*)d_in[b + o + 5];
  }

  // workspace layout
  char* ws = (char*)d_ws;
  size_t off = 0;
  auto take = [&](size_t bytes) -> size_t {
    size_t o = off;
    off += (bytes + 255) & ~(size_t)255;
    return o;
  };
  float*          h      = (float*)(ws + take((size_t)NN * HID * 4));
  unsigned short* hbf    = (unsigned short*)(ws + take((size_t)NN * HID * 2));
  float*          coords = (float*)(ws + take((size_t)NN * 2 * 4));
  float*          agg    = (float*)(ws + take((size_t)NN * 131 * 4)); // msg | xagg | cnt
  float* msg  = agg;
  float* xagg = agg + (size_t)NN * HID;
  float* cnt  = agg + (size_t)NN * 130;

  unsigned short *pk_ew1[3], *pk_ew2[3], *pk_cw1[2], *pk_nw1[3], *pk_nw2[3];
  const size_t PK256 = 8 * 8 * 512 * 2;  // K=256 packed bytes
  const size_t PK128 = 4 * 8 * 512 * 2;  // K=128 packed bytes
  for (int l = 0; l < NLAYERS; ++l) {
    pk_ew1[l] = (unsigned short*)(ws + take(PK256));
    pk_ew2[l] = (unsigned short*)(ws + take(PK128));
    if (l < 2) pk_cw1[l] = (unsigned short*)(ws + take(PK128));
    pk_nw1[l] = (unsigned short*)(ws + take(PK256));
    pk_nw2[l] = (unsigned short*)(ws + take(PK128));
  }

  auto pack = [&](const float* W, unsigned short* out, int K) {
    int total = (K / 32) * (HID / 16) * 512;
    pack_b_kernel<<<(total + 255) / 256, 256, 0, stream>>>(W, out, K, HID);
  };
  for (int l = 0; l < NLAYERS; ++l) {
    pack(ew1[l], pk_ew1[l], 256);  // rows 0..255; row 256 (dist) applied in epilogue
    pack(ew2[l], pk_ew2[l], 128);
    if (l < 2) pack(cw1[l], pk_cw1[l], 128);
    pack(nw1[l], pk_nw1[l], 256);
    pack(nw2[l], pk_nw2[l], 128);
  }

  proj_kernel<<<NN, 128, 0, stream>>>(x, proj_w, proj_b, h, hbf);
  copy_kernel<<<(NN * 2 + 255) / 256, 256, 0, stream>>>(pos, coords, NN * 2);

  const long long* erow = ei;
  const long long* ecol = ei + NE;

  for (int l = 0; l < NLAYERS; ++l) {
    const int upd = (l < NLAYERS - 1) ? 1 : 0;
    zero_kernel<<<(NN * 131 + 255) / 256, 256, 0, stream>>>(agg, NN * 131);
    egnn_edge_kernel<<<NE / 16, 128, 0, stream>>>(
        hbf, coords, erow, ecol,
        pk_ew1[l], eb1[l], ew1[l] + 256 * HID,
        pk_ew2[l], eb2[l],
        upd ? pk_cw1[l] : pk_ew2[l],
        upd ? cb1[l] : eb2[l],
        upd ? cw2[l] : eb2[l],
        upd ? cb2[l] : eb2[l],
        msg, xagg, cnt, upd);
    float* out = (l == NLAYERS - 1) ? (float*)d_out : h;
    egnn_node_kernel<<<NN / 16, 128, 0, stream>>>(
        h, hbf, msg, pk_nw1[l], nb1[l], pk_nw2[l], nb2[l], lng[l], lnb[l],
        out, hbf);
    if (upd)
      coord_update_kernel<<<(NN + 255) / 256, 256, 0, stream>>>(coords, xagg, cnt, NN);
  }
}
